// MultiLayerEdgeGAT_85813446574373
// MI455X (gfx1250) — matrix operand order repaired and checked by reading, not compiled, not run on hardware
//
#include <hip/hip_runtime.h>
#include <stdint.h>

#define N_NODES 50000
#define N_EDGES 800000
#define IN_F    64
#define EDGE_F  16
#define UNITS   128
#define NHEADS  8

#if __has_builtin(__builtin_amdgcn_global_load_async_to_lds_b128) && \
    __has_builtin(__builtin_amdgcn_s_wait_asynccnt)
#define USE_ASYNC_LDS 1
typedef int v4i_vs __attribute__((vector_size(16)));
typedef __attribute__((address_space(1))) v4i_vs* gv4i_p;
typedef __attribute__((address_space(3))) v4i_vs* lv4i_p;
#endif

typedef __attribute__((ext_vector_type(16))) __bf16 v16bf;
typedef __attribute__((ext_vector_type(8)))  float  v8f;

union FragBF {
  v16bf v;
  uint4 q[2];
  unsigned short u[16];
};

__device__ __forceinline__ unsigned short f2bf(float f) {
  unsigned u = __float_as_uint(f);
  unsigned r = 0x7FFFu + ((u >> 16) & 1u);   // round-to-nearest-even
  return (unsigned short)((u + r) >> 16);
}
__device__ __forceinline__ float bf2f(unsigned short h) {
  return __uint_as_float(((unsigned)h) << 16);
}
// order-preserving float->uint encoding for atomicMax-based segment_max
__device__ __forceinline__ unsigned encf(float f) {
  unsigned u = __float_as_uint(f);
  return (u & 0x80000000u) ? ~u : (u | 0x80000000u);
}
__device__ __forceinline__ float decf(unsigned k) {
  return __uint_as_float((k & 0x80000000u) ? (k & 0x7FFFFFFFu) : ~k);
}
#define ENC_NEG_INF 0x007FFFFFu

// ---------------------------------------------------------------------------
// Pack f32 weight (K x 128, row-major) into WMMA bf16 B-matrix lane layout.
// Tile (kt, nt): 32 lanes x 16 bf16 each, value j of lane L is
//   k = kt*32 + (L>>4)*8 + (j<8 ? j : j+8), col = nt*16 + (L&15). k >= Kreal zero-padded.
// ---------------------------------------------------------------------------
__global__ void pack_b_kernel(const float* __restrict__ W, int Kreal, int Ncols,
                              unsigned short* __restrict__ out) {
  int tile = blockIdx.x;
  int ntN = Ncols >> 4;
  int kt = tile / ntN, nt = tile % ntN;
  int lane = threadIdx.x;
  int col = nt * 16 + (lane & 15);
  int kb = kt * 32 + (lane >> 4) * 8;
  unsigned short* dst = out + (((size_t)tile * 32 + lane) << 4);
  for (int j = 0; j < 16; ++j) {
    int k = kb + (j < 8 ? j : j + 8);
    dst[j] = (k < Kreal) ? f2bf(W[(size_t)k * Ncols + col]) : (unsigned short)0;
  }
}

// Qe[k*H + h] = sum_j fc_edge[k, h*O + j] * ae[h, j]   (folds fe-GEMM away)
__global__ void qe_kernel(const float* __restrict__ fce, const float* __restrict__ ae,
                          float* __restrict__ Qe, int H, int O) {
  int i = blockIdx.x * 256 + threadIdx.x;
  if (i >= UNITS * H) return;
  int k = i / H, h = i % H;
  float s = 0.f;
  for (int j = 0; j < O; ++j) s += fce[(size_t)k * UNITS + h * O + j] * ae[h * O + j];
  Qe[k * H + h] = s;
}

__global__ void cvt_bf_kernel(const float* __restrict__ in, unsigned short* __restrict__ out,
                              int count) {
  int i = blockIdx.x * 256 + threadIdx.x;
  if (i < count) out[i] = f2bf(in[i]);
}

// ---------------------------------------------------------------------------
// Generic row-major bf16 GEMM: C(Mx128) = act( A(MxK) @ Bpk + bias ).
// grid.x = M/16, block = 256 (8 waves), wave w owns column tile w.
// A tile staged in LDS (async global->LDS when available); fragments follow
// the CDNA5 16-bit A / B VGPR layouts.
// ---------------------------------------------------------------------------
__global__ void gemm_bf16_kernel(const unsigned short* __restrict__ A, int K,
                                 const unsigned short* __restrict__ Bpk,
                                 const float* __restrict__ bias, int relu,
                                 float* __restrict__ Cf, unsigned short* __restrict__ Cb) {
  __shared__ __align__(16) unsigned short sA[16 * UNITS];
  const int tid = threadIdx.x;
  const int m0 = blockIdx.x * 16;
  const int cpr = K >> 3;                     // 16B chunks per row
#ifdef USE_ASYNC_LDS
  for (int i = tid; i < 16 * cpr; i += 256) {
    int r = i / cpr, c = i % cpr;
    __builtin_amdgcn_global_load_async_to_lds_b128(
        (gv4i_p)&A[(size_t)(m0 + r) * K + c * 8],
        (lv4i_p)&sA[r * K + c * 8], 0, 0);
  }
  __builtin_amdgcn_s_wait_asynccnt(0);
#else
  for (int i = tid; i < 16 * cpr; i += 256) {
    int r = i / cpr, c = i % cpr;
    *(uint4*)&sA[r * K + c * 8] = *(const uint4*)&A[(size_t)(m0 + r) * K + c * 8];
  }
#endif
  __syncthreads();

  const int lane = tid & 31, wv = tid >> 5;   // wv = column tile 0..7
  const int r = lane & 15, hi = lane >> 4;
  v8f acc = {};
  const int ksteps = K >> 5;
  for (int ks = 0; ks < ksteps; ++ks) {
    FragBF a, b;
    int k0 = ks * 32 + hi * 8;                // lanes 0-15: K 0-7/16-23; 16-31: K 8-15/24-31
    a.q[0] = *(const uint4*)&sA[r * K + k0];
    a.q[1] = *(const uint4*)&sA[r * K + k0 + 16];
    const uint4* bp = (const uint4*)(Bpk + (((size_t)(ks * 8 + wv) * 32 + lane) << 4));
    b.q[0] = bp[0];
    b.q[1] = bp[1];
    if (ks + 1 < ksteps)
      __builtin_prefetch((const void*)(Bpk + (((size_t)((ks + 1) * 8 + wv) * 32 + lane) << 4)), 0, 1);
    acc = __builtin_amdgcn_wmma_f32_16x16x32_bf16(false, a.v, false, b.v,
                                                  (short)0, acc, false, false);
  }
  const int cn = wv * 16 + r;
  float bv = bias ? bias[cn] : 0.f;
  for (int v = 0; v < 8; ++v) {
    int row = m0 + v + hi * 8;                // C/D layout: lanes>=16 hold rows +8
    float x = acc[v] + bv;
    if (relu) x = fmaxf(x, 0.f);
    if (Cf) Cf[(size_t)row * UNITS + cn] = x;
    if (Cb) Cb[(size_t)row * UNITS + cn] = f2bf(x);
  }
}

// wh = relu(w_in(E x 16) @ We + be), K padded 16->32 with zeros, bf16 out.
__global__ void edge_init_gemm(const float* __restrict__ Wei,
                               const unsigned short* __restrict__ Bpk,
                               const float* __restrict__ bias,
                               unsigned short* __restrict__ whb) {
  const int tid = threadIdx.x, lane = tid & 31, wv = tid >> 5;
  const int e0 = blockIdx.x * 16;
  const int r = lane & 15, hi = lane >> 4;
  FragBF a, b;
  const float* wr = Wei + (size_t)(e0 + r) * EDGE_F + hi * 8;
  for (int j = 0; j < 8; ++j) a.u[j] = f2bf(wr[j]);   // K 0-7 / 8-15
  for (int j = 8; j < 16; ++j) a.u[j] = 0;            // K 16-31 padding
  const uint4* bp = (const uint4*)(Bpk + (((size_t)wv * 32 + lane) << 4));
  b.q[0] = bp[0];
  b.q[1] = bp[1];
  v8f acc = {};
  acc = __builtin_amdgcn_wmma_f32_16x16x32_bf16(false, a.v, false, b.v,
                                                (short)0, acc, false, false);
  const int cn = wv * 16 + r;
  float bv = bias[cn];
  for (int v = 0; v < 8; ++v) {
    int row = e0 + v + hi * 8;
    float x = fmaxf(acc[v] + bv, 0.f);
    whb[(size_t)row * UNITS + cn] = f2bf(x);
  }
}

// wh = relu((wh + x[src] + x[dst]) @ Wemb + bemb), fused gather, in-place safe
// (each block reads only its own 16 rows into LDS before overwriting them).
__global__ void edge_update_gemm(unsigned short* __restrict__ whb,
                                 const unsigned short* __restrict__ xbf,
                                 const int* __restrict__ src, const int* __restrict__ dst,
                                 const unsigned short* __restrict__ Bpk,
                                 const float* __restrict__ bias) {
  __shared__ __align__(16) unsigned short sA[16 * UNITS];
  const int tid = threadIdx.x;
  const int e0 = blockIdx.x * 16;
  {
    int r = tid >> 4, cg = (tid & 15) * 8;
    int e = e0 + r;
    int s = src[e], d = dst[e];
    FragBF wq, xq, dq, o;
    wq.q[0] = *(const uint4*)&whb[(size_t)e * UNITS + cg];
    xq.q[0] = *(const uint4*)&xbf[(size_t)s * UNITS + cg];
    dq.q[0] = *(const uint4*)&xbf[(size_t)d * UNITS + cg];
    for (int j = 0; j < 8; ++j)
      o.u[j] = f2bf(bf2f(wq.u[j]) + bf2f(xq.u[j]) + bf2f(dq.u[j]));
    *(uint4*)&sA[r * UNITS + cg] = o.q[0];
  }
  __syncthreads();

  const int lane = tid & 31, wv = tid >> 5;
  const int r = lane & 15, hi = lane >> 4;
  v8f acc = {};
  for (int ks = 0; ks < 4; ++ks) {
    FragBF a, b;
    int k0 = ks * 32 + hi * 8;
    a.q[0] = *(const uint4*)&sA[r * UNITS + k0];
    a.q[1] = *(const uint4*)&sA[r * UNITS + k0 + 16];
    const uint4* bp = (const uint4*)(Bpk + (((size_t)(ks * 8 + wv) * 32 + lane) << 4));
    b.q[0] = bp[0];
    b.q[1] = bp[1];
    acc = __builtin_amdgcn_wmma_f32_16x16x32_bf16(false, a.v, false, b.v,
                                                  (short)0, acc, false, false);
  }
  const int cn = wv * 16 + r;
  float bv = bias[cn];
  for (int v = 0; v < 8; ++v) {
    int row = e0 + v + hi * 8;
    float x = fmaxf(acc[v] + bv, 0.f);
    whb[(size_t)row * UNITS + cn] = f2bf(x);
  }
}

// el/er: per-(node,head) dot of fs row-block with al/ar.
__global__ void attn_node_kernel(const float* __restrict__ fs,
                                 const float* __restrict__ al, const float* __restrict__ ar,
                                 float* __restrict__ el, float* __restrict__ er,
                                 int H, int O, int total) {
  int i = blockIdx.x * 256 + threadIdx.x;
  if (i >= total) return;
  int n = i / H, h = i % H;
  const float* p = fs + (size_t)n * UNITS + h * O;
  float sl = 0.f, sr = 0.f;
  for (int j = 0; j < O; ++j) { float f = p[j]; sl += f * al[h * O + j]; sr += f * ar[h * O + j]; }
  el[i] = sl;
  er[i] = sr;
}

__global__ void init_attn_kernel(unsigned int* __restrict__ emax, float* __restrict__ den,
                                 int count) {
  int i = blockIdx.x * 256 + threadIdx.x;
  if (i < count) { emax[i] = ENC_NEG_INF; den[i] = 0.f; }
}

// logit[e,h] = leaky( el[src] + er[dst] + wh[e,:] @ Qe[:,h] ); segment-max via
// monotone-encoded uint atomicMax into emax[dst,h].
__global__ void edge_logit_kernel(const unsigned short* __restrict__ whb,
                                  const float* __restrict__ Qe,
                                  const int* __restrict__ src, const int* __restrict__ dst,
                                  const float* __restrict__ el, const float* __restrict__ er,
                                  unsigned int* __restrict__ emax, float* __restrict__ ebuf,
                                  int H) {
  __shared__ float sQ[UNITS * NHEADS];
  const int tid = threadIdx.x;
  for (int i = tid; i < UNITS * H; i += 256) sQ[i] = Qe[i];
  __syncthreads();
  int e = blockIdx.x * 256 + tid;
  float acc[NHEADS];
  for (int h = 0; h < H; ++h) acc[h] = 0.f;
  for (int c = 0; c < UNITS; c += 8) {
    FragBF f;
    f.q[0] = *(const uint4*)&whb[(size_t)e * UNITS + c];
    for (int j = 0; j < 8; ++j) {
      float wv = bf2f(f.u[j]);
      for (int h = 0; h < H; ++h) acc[h] += wv * sQ[(c + j) * H + h];
    }
  }
  int s = src[e], d = dst[e];
  for (int h = 0; h < H; ++h) {
    float v = el[(size_t)s * H + h] + er[(size_t)d * H + h] + acc[h];
    v = (v > 0.f) ? v : 0.2f * v;            // leaky_relu(0.2)
    ebuf[(size_t)e * H + h] = v;
    atomicMax(&emax[(size_t)d * H + h], encf(v));
  }
}

// ex = exp(logit - emax[dst]); den[dst] += ex (in-place over ebuf).
__global__ void edge_exp_kernel(float* __restrict__ ebuf, const unsigned int* __restrict__ emax,
                                const int* __restrict__ dst, float* __restrict__ den, int H) {
  int e = blockIdx.x * 256 + threadIdx.x;
  int d = dst[e];
  for (int h = 0; h < H; ++h) {
    float m = decf(emax[(size_t)d * H + h]);
    float ex = __expf(ebuf[(size_t)e * H + h] - m);
    ebuf[(size_t)e * H + h] = ex;
    atomicAdd(&den[(size_t)d * H + h], ex);
  }
}

__global__ void rst_init_kernel(float* __restrict__ rst, const float* __restrict__ xh,
                                const float* __restrict__ bias) {
  int i = blockIdx.x * 256 + threadIdx.x;
  rst[i] = xh[i] + bias[i & (UNITS - 1)];    // identity residual + bias
}

// rst[dst] += fs[src] * alpha ; one wave per edge, lane covers 4 columns.
__global__ void edge_agg_kernel(const float* __restrict__ ebuf, const float* __restrict__ den,
                                const int* __restrict__ src, const int* __restrict__ dst,
                                const float* __restrict__ fs, float* __restrict__ rst, int H) {
  int tid = threadIdx.x, lane = tid & 31, wv = tid >> 5;
  int e = blockIdx.x * 8 + wv;
  int s = src[e], d = dst[e];
  int O = UNITS / H;
  for (int j = 0; j < 4; ++j) {
    int col = lane * 4 + j;
    int h = col / O;
    float dn = den[(size_t)d * H + h];
    float a = ebuf[(size_t)e * H + h] / (dn == 0.f ? 1.f : dn);
    atomicAdd(&rst[(size_t)d * UNITS + col], fs[(size_t)s * UNITS + col] * a);
  }
}

__global__ void finalize_kernel(const float* __restrict__ rst, int relu,
                                float* __restrict__ outf, unsigned short* __restrict__ outb) {
  int i = blockIdx.x * 256 + threadIdx.x;
  float v = rst[i];
  if (relu) v = fmaxf(v, 0.f);
  if (outf) outf[i] = v;
  if (outb) outb[i] = f2bf(v);
}

// ---------------------------------------------------------------------------
extern "C" void kernel_launch(void* const* d_in, const int* in_sizes, int n_in,
                              void* d_out, int out_size, void* d_ws, size_t ws_size,
                              hipStream_t stream) {
  (void)in_sizes; (void)n_in; (void)out_size; (void)ws_size;

  // Top-level dict insertion order: x, w, src, dst, params.
  // params flattened as a JAX pytree (dict keys sorted):
  //   We, Wemb[0..1], Wn, be, bemb[0..1], bn,
  //   layers[0..2] each sorted: ae, al, ar, bias, fc, fc_edge
  const float* x     = (const float*)d_in[0];
  const float* w     = (const float*)d_in[1];
  const int*   src   = (const int*)d_in[2];
  const int*   dst   = (const int*)d_in[3];
  const float* We    = (const float*)d_in[4];
  const float* Wemb[2] = { (const float*)d_in[5], (const float*)d_in[6] };
  const float* Wn    = (const float*)d_in[7];
  const float* be    = (const float*)d_in[8];
  const float* bemb[2] = { (const float*)d_in[9], (const float*)d_in[10] };
  const float* bn    = (const float*)d_in[11];
  const float *ae[3], *al[3], *ar[3], *lbias[3], *fc[3], *fce[3];
  for (int l = 0; l < 3; ++l) {
    ae[l]    = (const float*)d_in[12 + 6 * l + 0];
    al[l]    = (const float*)d_in[12 + 6 * l + 1];
    ar[l]    = (const float*)d_in[12 + 6 * l + 2];
    lbias[l] = (const float*)d_in[12 + 6 * l + 3];
    fc[l]    = (const float*)d_in[12 + 6 * l + 4];
    fce[l]   = (const float*)d_in[12 + 6 * l + 5];
  }

  // workspace carve-out (256B aligned)
  char* base = (char*)d_ws;
  size_t off = 0;
  auto take = [&](size_t bytes) -> char* {
    char* p = base + off;
    off = (off + bytes + 255) & ~(size_t)255;
    return p;
  };
  float*          xh     = (float*)take((size_t)N_NODES * UNITS * 4);
  float*          fsb    = (float*)take((size_t)N_NODES * UNITS * 4);
  float*          rst    = (float*)take((size_t)N_NODES * UNITS * 4);
  unsigned short* xh_bf  = (unsigned short*)take((size_t)N_NODES * UNITS * 2);
  unsigned short* xbf_in = (unsigned short*)take((size_t)N_NODES * IN_F * 2);
  unsigned short* whb    = (unsigned short*)take((size_t)N_EDGES * UNITS * 2);
  float*          ebuf   = (float*)take((size_t)N_EDGES * NHEADS * 4);
  unsigned int*   emax   = (unsigned int*)take((size_t)N_NODES * NHEADS * 4);
  float*          den    = (float*)take((size_t)N_NODES * NHEADS * 4);
  float*          elb    = (float*)take((size_t)N_NODES * NHEADS * 4);
  float*          erb    = (float*)take((size_t)N_NODES * NHEADS * 4);
  unsigned short* Wn_pk  = (unsigned short*)take((size_t)2 * 8 * 512 * 2);
  unsigned short* We_pk  = (unsigned short*)take((size_t)1 * 8 * 512 * 2);
  unsigned short* fc_pk[3];
  for (int l = 0; l < 3; ++l) fc_pk[l] = (unsigned short*)take((size_t)4 * 8 * 512 * 2);
  unsigned short* Wemb_pk[2];
  for (int l = 0; l < 2; ++l) Wemb_pk[l] = (unsigned short*)take((size_t)4 * 8 * 512 * 2);
  float* Qe[3];
  for (int l = 0; l < 3; ++l) Qe[l] = (float*)take((size_t)UNITS * NHEADS * 4);

  // weight prep (tiny)
  pack_b_kernel<<<2 * 8, 32, 0, stream>>>(Wn, IN_F, UNITS, Wn_pk);
  pack_b_kernel<<<1 * 8, 32, 0, stream>>>(We, EDGE_F, UNITS, We_pk);   // K padded 16->32
  for (int l = 0; l < 3; ++l)
    pack_b_kernel<<<4 * 8, 32, 0, stream>>>(fc[l], UNITS, UNITS, fc_pk[l]);
  for (int l = 0; l < 2; ++l)
    pack_b_kernel<<<4 * 8, 32, 0, stream>>>(Wemb[l], UNITS, UNITS, Wemb_pk[l]);
  for (int l = 0; l < 3; ++l) {
    int H = (l < 2) ? NHEADS : 1, O = UNITS / H;
    qe_kernel<<<(UNITS * H + 255) / 256, 256, 0, stream>>>(fce[l], ae[l], Qe[l], H, O);
  }

  // input projections
  cvt_bf_kernel<<<(N_NODES * IN_F + 255) / 256, 256, 0, stream>>>(x, xbf_in, N_NODES * IN_F);
  gemm_bf16_kernel<<<N_NODES / 16, 256, 0, stream>>>(xbf_in, IN_F, Wn_pk, bn, 1, xh, xh_bf);
  edge_init_gemm<<<N_EDGES / 16, 256, 0, stream>>>(w, We_pk, be, whb);

  // GAT layers
  for (int l = 0; l < 3; ++l) {
    int H = (l < 2) ? NHEADS : 1, O = UNITS / H;
    gemm_bf16_kernel<<<N_NODES / 16, 256, 0, stream>>>(xh_bf, UNITS, fc_pk[l], nullptr, 0,
                                                       fsb, nullptr);
    attn_node_kernel<<<(N_NODES * H + 255) / 256, 256, 0, stream>>>(fsb, al[l], ar[l],
                                                                    elb, erb, H, O, N_NODES * H);
    init_attn_kernel<<<(N_NODES * H + 255) / 256, 256, 0, stream>>>(emax, den, N_NODES * H);
    edge_logit_kernel<<<N_EDGES / 256, 256, 0, stream>>>(whb, Qe[l], src, dst, elb, erb,
                                                         emax, ebuf, H);
    edge_exp_kernel<<<N_EDGES / 256, 256, 0, stream>>>(ebuf, emax, dst, den, H);
    rst_init_kernel<<<N_NODES * UNITS / 256, 256, 0, stream>>>(rst, xh, lbias[l]);
    edge_agg_kernel<<<N_EDGES / 8, 256, 0, stream>>>(ebuf, den, src, dst, fsb, rst, H);
    if (l < 2) {
      finalize_kernel<<<N_NODES * UNITS / 256, 256, 0, stream>>>(rst, 1, xh, xh_bf);
      edge_update_gemm<<<N_EDGES / 16, 256, 0, stream>>>(whb, xh_bf, src, dst,
                                                         Wemb_pk[l], bemb[l]);
    } else {
      finalize_kernel<<<N_NODES * UNITS / 256, 256, 0, stream>>>(rst, 0, (float*)d_out, nullptr);
    }
  }
}
